// SNNLanguageModelLeaky_55576876810898
// MI455X (gfx1250) — matrix operand order repaired and checked by reading, not compiled, not run on hardware
//
#include <hip/hip_runtime.h>
#include <hip/hip_bf16.h>

// ---------------------------------------------------------------------------
// SNN language model for MI455X (gfx1250).
//
// Phase 0: Wout f32 -> bf16 in workspace (33MB, L2-resident; 192MB L2).
// Phase 1: sequential LIF recurrence. 16 blocks x 512 threads; block k owns
//          hidden units [32k,32k+32) for all 8 batches; W2/W3 row slices live
//          in LDS (131KB of 320KB, stride-513 padded -> bank-conflict free).
//          Spike vectors exchanged via L2; per-step broadcast into LDS done by
//          the Tensor Data Mover (tensor_load_to_lds + s_wait_tensorcnt) when
//          the builtin is available. Grid barrier = release/acquire atomics
//          (+ HW cluster barrier, NOP unless cluster-dispatched).
// Phase 2: logits = spk3 @ Wout^T + bout with v_wmma_f32_16x16x32_bf16.
//          Tiling: 8 waves/block, each wave = 4 M-tiles x 2 N-tiles
//          (64 rows x 256 cols per block, grid 125 x 64, no tails).
//          B-fragment reuse x4 => ~2.1GB L2 B-traffic + 0.5GB A-traffic,
//          approaching the 524MB output-store roofline (~22us @ 23.3TB/s).
// ---------------------------------------------------------------------------

typedef __bf16 bf16_t;
typedef bf16_t v8bf  __attribute__((ext_vector_type(8)));
typedef bf16_t v16bf __attribute__((ext_vector_type(16)));
typedef float  v8f   __attribute__((ext_vector_type(8)));

#define T_STEPS 512
#define BATCH   8
#define HID     512
#define VOCAB   32000
#define NBLK1   16
#define UNITS   32          // hidden units per phase-1 block
#define WPAD    513         // LDS row stride (odd vs 64 banks -> conflict free)
#define SPK_LDS_OFF (2u * UNITS * WPAD * 4u)   // byte offset of sSpk in LDS

// ---------------- phase 0: Wout f32 -> bf16 --------------------------------
__global__ __launch_bounds__(256) void cvt_bf16_kernel(const float* __restrict__ src,
                                                       __hip_bfloat16* __restrict__ dst,
                                                       int n) {
  int i = (blockIdx.x * 256 + threadIdx.x) * 4;
  if (i + 3 < n) {
    float4 v = *reinterpret_cast<const float4*>(src + i);
    dst[i + 0] = __float2bfloat16(v.x);
    dst[i + 1] = __float2bfloat16(v.y);
    dst[i + 2] = __float2bfloat16(v.z);
    dst[i + 3] = __float2bfloat16(v.w);
  }
}

// ---------------- TDM 1-D copy: global -> LDS ------------------------------
#if defined(__HIP_DEVICE_COMPILE__) && __has_builtin(__builtin_amdgcn_tensor_load_to_lds)
#define SNN_HAVE_TDM 1
typedef unsigned int u32x4 __attribute__((ext_vector_type(4)));
typedef int          i32x4 __attribute__((ext_vector_type(4)));
typedef int          i32x8 __attribute__((ext_vector_type(8)));

// D# per CDNA5 ISA: 1-D tile of `ndw` dwords.
__device__ __forceinline__ void tdm_load_1d(const void* gsrc, unsigned lds_off_bytes,
                                            unsigned ndw) {
  unsigned long long ga = (unsigned long long)(__SIZE_TYPE__)gsrc;
  u32x4 g0;
  g0[0] = 1u;                                                // count=1 (valid user D#)
  g0[1] = lds_off_bytes;                                     // lds_addr
  g0[2] = (unsigned)(ga & 0xFFFFFFFFu);                      // global_addr[31:0]
  g0[3] = (unsigned)((ga >> 32) & 0x01FFFFFFu) | (2u << 30); // global_addr[56:32]|type=2
  i32x8 g1;
  g1[0] = (int)(2u << 16);                  // data_size = 4B; no multicast/pad/iterate
  g1[1] = (int)((ndw & 0xFFFFu) << 16);     // tensor_dim0[15:0]  @ bits 63:48
  g1[2] = (int)(ndw >> 16);                 // tensor_dim0[31:16] @ bits 79:64
  g1[3] = (int)((ndw & 0xFFFFu) << 16);     // tile_dim0          @ bits 127:112
  g1[4] = 0;                                // tile_dim1/2 unused
  g1[5] = (int)ndw;                         // tensor_dim0_stride (low)
  g1[6] = 0;
  g1[7] = 0;
  i32x4 z4 = {0, 0, 0, 0};
#if defined(__clang_major__) && (__clang_major__ >= 23)
  i32x8 z8 = {0, 0, 0, 0, 0, 0, 0, 0};
  __builtin_amdgcn_tensor_load_to_lds(g0, g1, z4, z4, z8, 0);
#else
  __builtin_amdgcn_tensor_load_to_lds(g0, g1, z4, z4, 0);
#endif
}
#else
#define SNN_HAVE_TDM 0
#endif

// ---------------- grid-wide release/acquire barrier ------------------------
__device__ __forceinline__ void grid_sync(unsigned* cnt, unsigned target) {
  __threadfence();            // each thread publishes its global writes
  __syncthreads();
  if (threadIdx.x == 0) {
    // HW cluster barrier: fast path under cluster dispatch, S_NOP otherwise.
    __builtin_amdgcn_s_cluster_barrier();
    __hip_atomic_fetch_add(cnt, 1u, __ATOMIC_RELEASE, __HIP_MEMORY_SCOPE_AGENT);
    while (__hip_atomic_load(cnt, __ATOMIC_RELAXED, __HIP_MEMORY_SCOPE_AGENT) < target)
      __builtin_amdgcn_s_sleep(2);
    (void)__hip_atomic_load(cnt, __ATOMIC_ACQUIRE, __HIP_MEMORY_SCOPE_AGENT);
  }
  __syncthreads();
}

// ---------------- phase 1: sequential LIF recurrence -----------------------
__global__ __launch_bounds__(512) void snn_phase1_kernel(
    const int*   __restrict__ x,
    const float* __restrict__ emb,
    const float* __restrict__ pos_emb,
    const float* __restrict__ beta1,
    const float* __restrict__ beta2,
    const float* __restrict__ beta3,
    const float* __restrict__ W2, const float* __restrict__ b2,
    const float* __restrict__ W3, const float* __restrict__ b3,
    float* __restrict__ spkA,            // [8*512] exchange buffer (LIF1)
    float* __restrict__ spkB,            // [8*512] exchange buffer (LIF2)
    unsigned short* __restrict__ spk3_bf,// [4096*512] bf16 A-matrix for phase 2
    unsigned* __restrict__ cnt) {
  extern __shared__ float smem[];
  float* sW2  = smem;                       // UNITS * WPAD
  float* sW3  = sW2 + UNITS * WPAD;         // UNITS * WPAD
  float* sSpk = sW3 + UNITS * WPAD;         // BATCH * HID  (byte off SPK_LDS_OFF)
  float* sRed = sSpk + BATCH * HID;         // 512

  const int tid = threadIdx.x;
  const int b   = tid >> 6;                 // batch 0..7
  const int jl  = (tid >> 1) & 31;          // local hidden unit 0..31
  const int kh  = tid & 1;                  // K-half 0/1
  const int j0  = blockIdx.x * UNITS;
  const int j   = j0 + jl;

  // Preload this block's W2/W3 row slices into LDS (once for all 512 steps).
  for (int i = tid; i < UNITS * HID; i += 512) {
    int r = i >> 9, c = i & (HID - 1);
    sW2[r * WPAD + c] = W2[(size_t)(j0 + r) * HID + c];
    sW3[r * WPAD + c] = W3[(size_t)(j0 + r) * HID + c];
  }
  __syncthreads();

  const float bb1 = fminf(fmaxf(beta1[j], 0.f), 1.f);
  const float bb2 = fminf(fmaxf(beta2[j], 0.f), 1.f);
  const float bb3 = fminf(fmaxf(beta3[j], 0.f), 1.f);
  const float bi2 = b2[j];
  const float bi3 = b3[j];

  float m1 = 0.f, m2 = 0.f, m3 = 0.f;
  unsigned tgt = 0;
  const int k0 = kh * (HID / 2);
  const int ks = kh * 32;                   // bank-swizzle start

  for (int t = 0; t < T_STEPS; ++t) {
    // ---- LIF1 (owned by kh==0 threads) ----
    if (kh == 0) {
      int tok   = x[t * BATCH + b];
      float inp = emb[(size_t)tok * HID + j] + pos_emb[t * HID + j];
      float r1  = (m1 > 1.f) ? 1.f : 0.f;   // reset from previous membrane
      m1 = bb1 * m1 + inp - r1;
      float s1 = (m1 > 1.f) ? 1.f : 0.f;
      spkA[b * HID + j] = s1;
      if (t + 1 < T_STEPS)
        __builtin_prefetch(pos_emb + (t + 1) * HID + j, 0, 1);
    }
    tgt += NBLK1; grid_sync(cnt, tgt);

    // ---- broadcast spk1 into LDS (TDM when available), GEMM with W2 -------
#if SNN_HAVE_TDM
    if (tid < 32) {                          // wave 0 issues the DMA
      tdm_load_1d(spkA, SPK_LDS_OFF, BATCH * HID);
      __builtin_amdgcn_s_wait_tensorcnt(0);
    }
#else
    for (int i = tid; i < BATCH * HID; i += 512) sSpk[i] = spkA[i];
#endif
    __syncthreads();
    {
      const float* w  = sW2 + jl * WPAD;
      const float* sp = sSpk + b * HID;
      float acc = 0.f;
      #pragma unroll 8
      for (int k = k0 + ks; k < k0 + HID / 2; ++k) acc = fmaf(sp[k], w[k], acc);
      for (int k = k0; k < k0 + ks; ++k)           acc = fmaf(sp[k], w[k], acc);
      sRed[tid] = acc;
    }
    __syncthreads();
    if (kh == 0) {
      float h = fmaxf(sRed[tid] + sRed[tid + 1] + bi2, 0.f);
      float r2 = (m2 > 1.f) ? 1.f : 0.f;
      m2 = bb2 * m2 + h - r2;
      float s2 = (m2 > 1.f) ? 1.f : 0.f;
      spkB[b * HID + j] = s2;
    }
    tgt += NBLK1; grid_sync(cnt, tgt);

    // ---- broadcast spk2 into LDS, GEMM with W3 ----------------------------
#if SNN_HAVE_TDM
    if (tid < 32) {
      tdm_load_1d(spkB, SPK_LDS_OFF, BATCH * HID);
      __builtin_amdgcn_s_wait_tensorcnt(0);
    }
#else
    for (int i = tid; i < BATCH * HID; i += 512) sSpk[i] = spkB[i];
#endif
    __syncthreads();
    {
      const float* w  = sW3 + jl * WPAD;
      const float* sp = sSpk + b * HID;
      float acc = 0.f;
      #pragma unroll 8
      for (int k = k0 + ks; k < k0 + HID / 2; ++k) acc = fmaf(sp[k], w[k], acc);
      for (int k = k0; k < k0 + ks; ++k)           acc = fmaf(sp[k], w[k], acc);
      sRed[tid] = acc;
    }
    __syncthreads();
    if (kh == 0) {
      float h = fmaxf(sRed[tid] + sRed[tid + 1] + bi3, 0.f);
      float r3 = (m3 > 1.f) ? 1.f : 0.f;
      m3 = bb3 * m3 + h - r3;
      float s3 = (m3 > 1.f) ? 1.f : 0.f;
      spk3_bf[(size_t)(t * BATCH + b) * HID + j] = s3 > 0.f ? 0x3F80u : 0u;
    }
  }
}

// ---------------- phase 2: logits = spk3 @ Wout^T + bout (WMMA) ------------
__device__ __forceinline__ v16bf ld_frag(const bf16_t* p0, const bf16_t* p1) {
  v8bf a = *reinterpret_cast<const v8bf*>(p0);
  v8bf b = *reinterpret_cast<const v8bf*>(p1);
  return __builtin_shufflevector(a, b, 0, 1, 2, 3, 4, 5, 6, 7,
                                       8, 9, 10, 11, 12, 13, 14, 15);
}

__global__ __launch_bounds__(256) void snn_phase2_kernel(
    const bf16_t* __restrict__ A,     // [4096][512] spk3, bf16
    const bf16_t* __restrict__ Wb,    // [32000][512] Wout, bf16 (L2-resident)
    const float*  __restrict__ bout,  // [32000]
    float* __restrict__ out) {        // [4096][32000]
  const int l  = threadIdx.x & 31;
  const int wv = threadIdx.x >> 5;              // wave 0..7 = N subgroup
  const int r0 = blockIdx.y * 64;               // 4 M-tiles of 16 rows
  const int n0 = blockIdx.x * 256 + wv * 32;    // 2 N-tiles of 16 cols

  // A operand, contiguous k order chosen here (k = 16h + s): lane l holds row (l&15); 16 contiguous K at (l>=16?16:0).
  const int kA = (l >> 4) * 16;
  // B operand, same contiguous k order as A: lane l holds col (l&15); 16 contiguous K at (l>=16?16:0).
  const int colt = l & 15;
  const int kB   = (l >> 4) * 16;

  const bf16_t* Arow[4];
  #pragma unroll
  for (int mt = 0; mt < 4; ++mt)
    Arow[mt] = A + (size_t)(r0 + mt * 16 + (l & 15)) * HID + kA;
  const bf16_t* Brow[2];
  #pragma unroll
  for (int nt = 0; nt < 2; ++nt)
    Brow[nt] = Wb + (size_t)(n0 + nt * 16 + colt) * HID + kB;

  v8f acc[4][2];
  #pragma unroll
  for (int mt = 0; mt < 4; ++mt)
    #pragma unroll
    for (int nt = 0; nt < 2; ++nt)
      acc[mt][nt] = (v8f){0, 0, 0, 0, 0, 0, 0, 0};

  #pragma unroll
  for (int kb = 0; kb < HID; kb += 32) {
    v16bf bf0 = ld_frag(Brow[0] + kb, Brow[0] + kb + 8);
    v16bf bf1 = ld_frag(Brow[1] + kb, Brow[1] + kb + 8);
    #pragma unroll
    for (int mt = 0; mt < 4; ++mt) {
      v16bf af = ld_frag(Arow[mt] + kb, Arow[mt] + kb + 8);
      acc[mt][0] = __builtin_amdgcn_wmma_f32_16x16x32_bf16(false, af, false, bf0,
                                                           (short)0, acc[mt][0], false, false);
      acc[mt][1] = __builtin_amdgcn_wmma_f32_16x16x32_bf16(false, af, false, bf1,
                                                           (short)0, acc[mt][1], false, false);
    }
  }

  // f32 C/D layout: lane l holds col (l&15); VGPR i -> row i + 8*(l>=16).
  #pragma unroll
  for (int nt = 0; nt < 2; ++nt) {
    int cg = n0 + nt * 16 + colt;
    float bi = bout[cg];
    #pragma unroll
    for (int mt = 0; mt < 4; ++mt) {
      int rbase = r0 + mt * 16 + ((l >> 4) << 3);
      #pragma unroll
      for (int i = 0; i < 8; ++i)
        out[(size_t)(rbase + i) * VOCAB + cg] = acc[mt][nt][i] + bi;
    }
  }
}

// ---------------------------------------------------------------------------
extern "C" void kernel_launch(void* const* d_in, const int* in_sizes, int n_in,
                              void* d_out, int out_size, void* d_ws, size_t ws_size,
                              hipStream_t stream) {
  const int*   x       = (const int*)  d_in[0];
  const float* emb     = (const float*)d_in[1];
  const float* pos_emb = (const float*)d_in[2];
  const float* beta1   = (const float*)d_in[3];
  const float* beta2   = (const float*)d_in[4];
  const float* beta3   = (const float*)d_in[5];
  const float* W2      = (const float*)d_in[6];
  const float* b2      = (const float*)d_in[7];
  const float* W3      = (const float*)d_in[8];
  const float* b3      = (const float*)d_in[9];
  const float* Wout    = (const float*)d_in[10];
  const float* bout    = (const float*)d_in[11];
  float* out = (float*)d_out;

  // workspace layout (~37MB total)
  char* w = (char*)d_ws;
  unsigned*       cnt   = (unsigned*)w;                              // [0, 4K)
  float*          spkA  = (float*)(w + 4096);                        // 16KB
  float*          spkB  = (float*)(w + 4096 + 16384);                // 16KB
  unsigned short* A3    = (unsigned short*)(w + 65536);              // 4096*512*2  = 4MB
  unsigned short* Wbf   = (unsigned short*)(w + 65536 + 4194304);    // 32000*512*2 = 32.8MB

  // zero the barrier counters every launch (ws is not re-poisoned by harness)
  hipMemsetAsync(w, 0, 4096, stream);

  // phase 0: Wout -> bf16
  cvt_bf16_kernel<<<(VOCAB * HID) / (256 * 4), 256, 0, stream>>>(
      Wout, (__hip_bfloat16*)Wbf, VOCAB * HID);

  // phase 1: recurrence (150KB LDS per block -> opt in to large dynamic LDS)
  const int smem_bytes = (2 * UNITS * WPAD + BATCH * HID + 512) * (int)sizeof(float);
  (void)hipFuncSetAttribute((const void*)snn_phase1_kernel,
                            hipFuncAttributeMaxDynamicSharedMemorySize, 160 * 1024);
  snn_phase1_kernel<<<NBLK1, 512, smem_bytes, stream>>>(
      x, emb, pos_emb, beta1, beta2, beta3, W2, b2, W3, b3,
      spkA, spkB, A3, cnt);

  // phase 2: WMMA vocab GEMM, parallel over all (t, b); 125*256 = 32000,
  // 64*64 = 4096 -> no tail handling needed.
  dim3 g2(VOCAB / 256, (T_STEPS * BATCH) / 64);
  snn_phase2_kernel<<<g2, 256, 0, stream>>>(
      (const bf16_t*)A3, (const bf16_t*)Wbf, bout, out);
}
